// LSTMCustom_15693810500357
// MI455X (gfx1250) — matrix operand order repaired and checked by reading, not compiled, not run on hardware
//
#include <hip/hip_runtime.h>

// ---------------------------------------------------------------------------
// LSTM cell, B=8192 E=512 H=1024.
//   g = [xt|h] @ [Wx|Wh]^T + (bx+bh)      (M=8192, N=4096, K=1536)
//   i,f,o = sigmoid(g_i,f,o); cc = tanh(g_c)
//   c_new = f*c + i*cc ; h_new = o*c_new
// Strategy: split-bf16 (hi/lo) WMMA GEMM (3x v_wmma_f32_16x16x32_bf16 per
// tile-pair ~ fp32 accuracy), async-LDS-staged weight tiles, fused epilogue.
// ---------------------------------------------------------------------------

#define DEV static __device__ __forceinline__

typedef __attribute__((ext_vector_type(16))) __bf16        v16bf;
typedef __attribute__((ext_vector_type(8)))  float         v8f;
typedef __attribute__((ext_vector_type(4)))  unsigned int  u32x4;

static constexpr int Bsz = 8192;
static constexpr int E   = 512;
static constexpr int H   = 1024;
static constexpr int K   = E + H;     // 1536
static constexpr int NG  = 4 * H;     // 4096 fused gate rows of W

// ---- CDNA5 async global->LDS path (guarded; falls back to sync staging) ---
#if defined(__has_builtin)
#  if __has_builtin(__builtin_amdgcn_global_load_async_to_lds_b128)
#    define USE_ASYNC_LDS 1
#  endif
#  if __has_builtin(__builtin_amdgcn_s_wait_asynccnt)
#    define HAVE_WAIT_ASYNC 1
#  endif
#endif

#ifdef USE_ASYNC_LDS
// builtin proto (from clang diagnostic): param0 = 'int __vector(4) __device__ *'
// i.e. pointer-to-int4 in AS(1); dst is the LDS (AS(3)) counterpart.
typedef int i32x4_vs __attribute__((vector_size(16)));
typedef __attribute__((address_space(1))) i32x4_vs gvec_t;
typedef __attribute__((address_space(3))) i32x4_vs lvec_t;

DEV void cp_async16(const unsigned short* g, unsigned short* l) {
    gvec_t* gp = reinterpret_cast<gvec_t*>((unsigned long long)(uintptr_t)g);
    lvec_t* lp = reinterpret_cast<lvec_t*>((unsigned)(uintptr_t)l);
    __builtin_amdgcn_global_load_async_to_lds_b128(gp, lp, 0, 0);
}
DEV void wait_async0() {
#ifdef HAVE_WAIT_ASYNC
    __builtin_amdgcn_s_wait_asynccnt(0);
#else
    asm volatile("s_wait_asynccnt 0x0" ::: "memory");
#endif
}
#endif

// ---- round-to-nearest-even fp32 -> bf16 split: x = hi + lo ---------------
DEV unsigned short f32_to_bf16_rne(float x) {
    unsigned u = __float_as_uint(x);
    unsigned r = u + 0x7FFFu + ((u >> 16) & 1u);
    return (unsigned short)(r >> 16);
}
DEV void split_bf16(float x, unsigned short& hi, unsigned short& lo) {
    unsigned short h = f32_to_bf16_rne(x);
    float hf = __uint_as_float(((unsigned)h) << 16);
    hi = h;
    lo = f32_to_bf16_rne(x - hf);
}

// ---- fast gate activations (v_exp_f32 + v_rcp_f32, correct at +-inf) ------
DEV float fast_sigmoid(float x) {
    return __builtin_amdgcn_rcpf(1.0f + __expf(-x));
}
DEV float fast_tanh(float x) {
    return 1.0f - 2.0f * __builtin_amdgcn_rcpf(1.0f + __expf(2.0f * x));
}

// ---- prologue: build bf16 hi/lo planes of A = [xt|h]  (B x K) -------------
__global__ void lstm_cvt_A(const float* __restrict__ xt,
                           const float* __restrict__ h,
                           unsigned short* __restrict__ Ahi,
                           unsigned short* __restrict__ Alo) {
    int k = blockIdx.x * 256 + threadIdx.x;   // 0..K-1   (gridDim.x = K/256)
    int b = blockIdx.y;                       // 0..B-1
    float v = (k < E) ? xt[(size_t)b * E + k]
                      : h [(size_t)b * H + (k - E)];
    unsigned short hi, lo;
    split_bf16(v, hi, lo);
    size_t idx = (size_t)b * K + k;
    Ahi[idx] = hi; Alo[idx] = lo;
}

// ---- prologue: build bf16 hi/lo planes of W = [Wx|Wh] gate-major (NG x K) -
__global__ void lstm_cvt_W(const float* __restrict__ Wix, const float* __restrict__ Wfx,
                           const float* __restrict__ Wox, const float* __restrict__ Wcx,
                           const float* __restrict__ Wih, const float* __restrict__ Wfh,
                           const float* __restrict__ Woh, const float* __restrict__ Wch,
                           unsigned short* __restrict__ Whi,
                           unsigned short* __restrict__ Wlo) {
    int k = blockIdx.x * 256 + threadIdx.x;   // 0..K-1
    int n = blockIdx.y;                       // 0..NG-1
    int gate = n >> 10;
    int r    = n & 1023;
    const float* Wx = (gate == 0) ? Wix : (gate == 1) ? Wfx : (gate == 2) ? Wox : Wcx;
    const float* Wh = (gate == 0) ? Wih : (gate == 1) ? Wfh : (gate == 2) ? Woh : Wch;
    float v = (k < E) ? Wx[(size_t)r * E + k]
                      : Wh[(size_t)r * H + (k - E)];
    unsigned short hi, lo;
    split_bf16(v, hi, lo);
    size_t idx = (size_t)n * K + k;
    Whi[idx] = hi; Wlo[idx] = lo;
}

// ---- prologue: fused bias bx+bh, gate-major [NG] --------------------------
__global__ void lstm_cvt_bias(const float* __restrict__ bix, const float* __restrict__ bfx,
                              const float* __restrict__ box, const float* __restrict__ bcx,
                              const float* __restrict__ bih, const float* __restrict__ bfh,
                              const float* __restrict__ boh, const float* __restrict__ bch,
                              float* __restrict__ biasC) {
    int n = blockIdx.x * 256 + threadIdx.x;
    if (n >= NG) return;
    int gate = n >> 10, r = n & 1023;
    const float* bx = (gate == 0) ? bix : (gate == 1) ? bfx : (gate == 2) ? box : bcx;
    const float* bh = (gate == 0) ? bih : (gate == 1) ? bfh : (gate == 2) ? boh : bch;
    biasC[n] = bx[r] + bh[r];
}

// ---- fragment loads -------------------------------------------------------
// Build a 16-element bf16 fragment from two 16-byte chunks.
DEV v16bf ldfrag(const unsigned short* p0, const unsigned short* p1) {
    u32x4 t0 = *reinterpret_cast<const u32x4*>(p0);
    u32x4 t1 = *reinterpret_cast<const u32x4*>(p1);
    struct { u32x4 a, b; } raw = { t0, t1 };
    v16bf v;
    __builtin_memcpy(&v, &raw, sizeof(v));
    return v;
}

DEV v8f wmma_bf16(v16bf a, v16bf b, v8f c) {
    return __builtin_amdgcn_wmma_f32_16x16x32_bf16(
        /*neg_a=*/false, a, /*neg_b=*/false, b,
        /*c_mod=*/(short)0, c, /*reuse_a=*/false, /*reuse_b=*/false);
}

// ---------------------------------------------------------------------------
// Main fused kernel. Grid: (H/16, B/256), block 256 (8 waves).
// Wave w: rows [m0+32w, +32), cols [n0,n0+16) for all 4 gates.
// ---------------------------------------------------------------------------
__global__ void __launch_bounds__(256)
lstm_wmma_gemm(const unsigned short* __restrict__ Ahi,
               const unsigned short* __restrict__ Alo,
               const unsigned short* __restrict__ Whi,
               const unsigned short* __restrict__ Wlo,
               const float* __restrict__ biasC,
               const float* __restrict__ Cin,
               float* __restrict__ Out) {
    __shared__ alignas(16) unsigned short sWhi[2][64 * 32];   // 64 rows x 32 k-elems
    __shared__ alignas(16) unsigned short sWlo[2][64 * 32];

    const int tid    = threadIdx.x;
    const int lane   = tid & 31;
    const int wave   = tid >> 5;
    const int laneLo = lane & 15;
    const int half   = lane >> 4;

    const int n0     = blockIdx.x * 16;        // column within each gate
    const int m_wg   = blockIdx.y * 256;
    const int m_wave = m_wg + wave * 32;

    // A row base pointers (wave-private, 2 row-tiles of 16, hi & lo planes)
    const unsigned short* pA0h = Ahi + (size_t)(m_wave + laneLo) * K;
    const unsigned short* pA1h = pA0h + (size_t)16 * K;
    const unsigned short* pA0l = Alo + (size_t)(m_wave + laneLo) * K;
    const unsigned short* pA1l = pA0l + (size_t)16 * K;

    // cooperative W tile staging: 64 rows (4 gates x 16 cols) x 32 k-elems
    const int srow   = tid >> 2;               // 0..63
    const int schunk = tid & 3;                // 0..3 (8 elems each)
    const int sgrow  = (srow >> 4) * 1024 + n0 + (srow & 15);
    const size_t gW  = (size_t)sgrow * K + schunk * 8;
    const int    lW  = srow * 32 + schunk * 8;

    v8f acc[4][2];
#pragma unroll
    for (int g = 0; g < 4; ++g) { acc[g][0] = {}; acc[g][1] = {}; }

    // prime buffer 0
#ifdef USE_ASYNC_LDS
    cp_async16(Whi + gW, &sWhi[0][lW]);
    cp_async16(Wlo + gW, &sWlo[0][lW]);
    wait_async0();
#else
    *reinterpret_cast<u32x4*>(&sWhi[0][lW]) = *reinterpret_cast<const u32x4*>(Whi + gW);
    *reinterpret_cast<u32x4*>(&sWlo[0][lW]) = *reinterpret_cast<const u32x4*>(Wlo + gW);
#endif
    __syncthreads();

    int buf = 0;
#pragma unroll 1
    for (int k0 = 0; k0 < K; k0 += 32, buf ^= 1) {
        // stage next W tile into the other buffer
        if (k0 + 32 < K) {
            size_t g2 = gW + (size_t)(k0 + 32);
#ifdef USE_ASYNC_LDS
            cp_async16(Whi + g2, &sWhi[buf ^ 1][lW]);
            cp_async16(Wlo + g2, &sWlo[buf ^ 1][lW]);
#else
            *reinterpret_cast<u32x4*>(&sWhi[buf ^ 1][lW]) =
                *reinterpret_cast<const u32x4*>(Whi + g2);
            *reinterpret_cast<u32x4*>(&sWlo[buf ^ 1][lW]) =
                *reinterpret_cast<const u32x4*>(Wlo + g2);
#endif
        }

        // A fragments (16x32 bf16 layout: 8-elem half-wave interleave)
        const int offA = k0 + half * 8;
        v16bf a0h = ldfrag(pA0h + offA, pA0h + offA + 16);
        v16bf a0l = ldfrag(pA0l + offA, pA0l + offA + 16);
        v16bf a1h = ldfrag(pA1h + offA, pA1h + offA + 16);
        v16bf a1l = ldfrag(pA1l + offA, pA1l + offA + 16);

        // B fragments from LDS (32x16 bf16 layout: 8-elem half-wave interleave)
#pragma unroll
        for (int g = 0; g < 4; ++g) {
            const int offB = (g * 16 + laneLo) * 32 + half * 8;
            v16bf bh = ldfrag(&sWhi[buf][offB], &sWhi[buf][offB + 16]);
            v16bf bl = ldfrag(&sWlo[buf][offB], &sWlo[buf][offB + 16]);

            acc[g][0] = wmma_bf16(a0h, bh, acc[g][0]);
            acc[g][0] = wmma_bf16(a0h, bl, acc[g][0]);
            acc[g][0] = wmma_bf16(a0l, bh, acc[g][0]);

            acc[g][1] = wmma_bf16(a1h, bh, acc[g][1]);
            acc[g][1] = wmma_bf16(a1h, bl, acc[g][1]);
            acc[g][1] = wmma_bf16(a1l, bh, acc[g][1]);
        }

#ifdef USE_ASYNC_LDS
        wait_async0();      // next tile fully resident in LDS
#endif
        __syncthreads();
    }

    // ---- fused LSTM epilogue ----------------------------------------------
    const int hcol = n0 + laneLo;
    const float bi = biasC[0 * 1024 + hcol];
    const float bf = biasC[1 * 1024 + hcol];
    const float bo = biasC[2 * 1024 + hcol];
    const float bc = biasC[3 * 1024 + hcol];

#pragma unroll
    for (int t = 0; t < 2; ++t) {
#pragma unroll
        for (int r = 0; r < 8; ++r) {
            const int m = m_wave + 16 * t + 8 * half + r;
            const size_t idx = (size_t)m * H + hcol;

            float ig = fast_sigmoid(acc[0][t][r] + bi);
            float fg = fast_sigmoid(acc[1][t][r] + bf);
            float og = fast_sigmoid(acc[2][t][r] + bo);
            float cg = fast_tanh   (acc[3][t][r] + bc);

            float cn = fg * Cin[idx] + ig * cg;
            Out[idx] = og * cn;                      // h_new
            Out[(size_t)Bsz * H + idx] = cn;         // c_new
        }
    }
}

// ---------------------------------------------------------------------------
extern "C" void kernel_launch(void* const* d_in, const int* in_sizes, int n_in,
                              void* d_out, int out_size, void* d_ws, size_t ws_size,
                              hipStream_t stream) {
    const float* xt  = (const float*)d_in[0];
    const float* h   = (const float*)d_in[1];
    const float* c   = (const float*)d_in[2];
    const float* Wix = (const float*)d_in[3];  const float* bix = (const float*)d_in[4];
    const float* Wfx = (const float*)d_in[5];  const float* bfx = (const float*)d_in[6];
    const float* Wox = (const float*)d_in[7];  const float* box = (const float*)d_in[8];
    const float* Wcx = (const float*)d_in[9];  const float* bcx = (const float*)d_in[10];
    const float* Wih = (const float*)d_in[11]; const float* bih = (const float*)d_in[12];
    const float* Wfh = (const float*)d_in[13]; const float* bfh = (const float*)d_in[14];
    const float* Woh = (const float*)d_in[15]; const float* boh = (const float*)d_in[16];
    const float* Wch = (const float*)d_in[17]; const float* bch = (const float*)d_in[18];
    float* out = (float*)d_out;

    // workspace partition (bf16 planes + fused bias)
    unsigned short* Ahi = (unsigned short*)d_ws;
    unsigned short* Alo = Ahi + (size_t)Bsz * K;
    unsigned short* Whi = Alo + (size_t)Bsz * K;
    unsigned short* Wlo = Whi + (size_t)NG * K;
    float*        biasC = (float*)(Wlo + (size_t)NG * K);

    lstm_cvt_A<<<dim3(K / 256, Bsz), 256, 0, stream>>>(xt, h, Ahi, Alo);
    lstm_cvt_W<<<dim3(K / 256, NG), 256, 0, stream>>>(Wix, Wfx, Wox, Wcx,
                                                      Wih, Wfh, Woh, Wch, Whi, Wlo);
    lstm_cvt_bias<<<dim3(NG / 256), 256, 0, stream>>>(bix, bfx, box, bcx,
                                                      bih, bfh, boh, bch, biasC);

    lstm_wmma_gemm<<<dim3(H / 16, Bsz / 256), 256, 0, stream>>>(
        Ahi, Alo, Whi, Wlo, biasC, c, out);
}
